// DMFNSAttentionHead_54408645706196
// MI455X (gfx1250) — matrix operand order repaired and checked by reading, not compiled, not run on hardware
//
#include <hip/hip_runtime.h>
#include <hip/hip_bf16.h>

typedef __attribute__((ext_vector_type(16))) _Float16 v16h;
typedef __attribute__((ext_vector_type(8)))  _Float16 v8h;
typedef __attribute__((ext_vector_type(8)))  float    v8f;

typedef unsigned int       u32;
typedef unsigned long long u64;
typedef u32 __attribute__((ext_vector_type(4))) u32x4;
typedef int __attribute__((ext_vector_type(4))) i32x4;
typedef int __attribute__((ext_vector_type(8))) i32x8;

namespace {
constexpr int   B      = 4;
constexpr int   S      = 2048;
constexpr int   D      = 512;
constexpr int   HD     = 512;
constexpr float EPS    = 1e-12f;
constexpr float INV_BW = 1.0f / 512.0f;   // alpha=2 -> score = exp(-sq/BW)
}

// ---------------------------------------------------------------------------
// WMMA tile loaders (wave32, 16x16x32 f16).
// A-matrix (MxK=16x32): lane L -> row M=L%16; lanes 0-15 hold K=0..7 / 16..23,
// lanes 16-31 hold K=8..15 / 24..31.
// ---------------------------------------------------------------------------
__device__ __forceinline__ v16h load_a_f16(const _Float16* __restrict__ base,
                                           int stride, int lane) {
  const _Float16* p = base + (size_t)(lane & 15) * stride + ((lane >> 4) << 3);
  union { v16h v; v8h h[2]; } u;
  u.h[0] = *(const v8h*)(p);
  u.h[1] = *(const v8h*)(p + 16);
  return u.v;
}

// A-operand from an f32 source (probs), converted to f16 on load.
__device__ __forceinline__ v16h load_a_f32(const float* __restrict__ base,
                                           int stride, int lane) {
  const float* p = base + (size_t)(lane & 15) * stride + ((lane >> 4) << 3);
  v8f f0 = *(const v8f*)(p);
  v8f f1 = *(const v8f*)(p + 16);
  v16h r;
#pragma unroll
  for (int i = 0; i < 8; ++i) {
    r[i]     = (_Float16)f0[i];
    r[i + 8] = (_Float16)f1[i];
  }
  return r;
}

// B-matrix (KxN=32x16): lane L -> col N=L%16; lanes 0-15 hold K=0..7 / 16..23,
// lanes 16-31 hold K=8..15 / 24..31.
__device__ __forceinline__ v16h load_b_f16(const _Float16* __restrict__ base,
                                           int stride, int lane) {
  const _Float16* p = base + (size_t)(lane & 15) * stride + ((lane >> 4) << 3);
  union { v16h v; v8h h[2]; } u;
  u.h[0] = *(const v8h*)(p);
  u.h[1] = *(const v8h*)(p + 16);
  return u.v;
}

__device__ __forceinline__ v8f wmma_f16(v16h a, v16h b, v8f c) {
  return __builtin_amdgcn_wmma_f32_16x16x32_f16(false, a, false, b,
                                                (short)0, c, false, false);
}

// ---------------------------------------------------------------------------
// Tensor Data Mover: 2D f16 tile load global->LDS.
// D# per CDNA5 ISA ch.8: group0 = {count/type/lds_addr/global_addr},
// group1 = {data_size, tensor dims, tile dims, strides}. 2-D tile so groups
// 2/3 are zero. data_size=1 (2 bytes).
// ---------------------------------------------------------------------------
__device__ __forceinline__ void tdm_load_2d_f16(u32 lds_byte_addr, u64 gaddr,
                                                u32 tensor_d0, u32 tensor_d1,
                                                u32 tile_d0, u32 tile_d1,
                                                u32 stride0) {
  u32x4 g0;
  g0.x = 1u;                                             // count=1 (valid D#)
  g0.y = lds_byte_addr;                                  // lds_addr [63:32]
  g0.z = (u32)(gaddr & 0xffffffffu);                     // global_addr lo
  g0.w = (u32)((gaddr >> 32) & 0x1ffffffu) | (2u << 30); // addr hi | type=2
  i32x8 g1;
  g1[0] = (int)(1u << 16);                               // data_size=2B
  g1[1] = (int)((tensor_d0 & 0xffffu) << 16);            // tensor_dim0 lo16
  g1[2] = (int)((tensor_d0 >> 16) | ((tensor_d1 & 0xffffu) << 16));
  g1[3] = (int)((tensor_d1 >> 16) | (tile_d0 << 16));    // tile_dim0
  g1[4] = (int)(tile_d1);                                // tile_dim1 (dim2=0)
  g1[5] = (int)stride0;                                  // tensor_dim0_stride
  g1[6] = 0;
  g1[7] = 0;
  i32x4 z4 = {0, 0, 0, 0};
#if __clang_major__ >= 23
  i32x8 z8 = {0, 0, 0, 0, 0, 0, 0, 0};
  __builtin_amdgcn_tensor_load_to_lds(g0, g1, z4, z4, z8, 0);
#else
  __builtin_amdgcn_tensor_load_to_lds(g0, g1, z4, z4, 0);
#endif
}

// ---------------------------------------------------------------------------
// Utility kernels
// ---------------------------------------------------------------------------
__global__ __launch_bounds__(256) void cvt_f32_f16_kernel(
    const float* __restrict__ in, _Float16* __restrict__ out, int n) {
  int i = blockIdx.x * 256 + threadIdx.x;
  if (i < n) out[i] = (_Float16)in[i];
}

__global__ __launch_bounds__(256) void zero_f32_kernel(
    float* __restrict__ p, int n) {
  int i = blockIdx.x * 256 + threadIdx.x;
  if (i < n) p[i] = 0.0f;
}

// ---------------------------------------------------------------------------
// Projections: out(m,n) = sum_k x(m,k) * W(n,k) + bias(n).
// MODE 0: f16 row-major (q,k).  MODE 1: f16 transposed per batch ([b][d][t]).
// ---------------------------------------------------------------------------
template <int MODE>
__global__ __launch_bounds__(256) void proj_kernel(
    const _Float16* __restrict__ xh, const _Float16* __restrict__ Wh,
    const float* __restrict__ bias, _Float16* __restrict__ outh) {
  const int lane = threadIdx.x & 31;
  const int wave = threadIdx.x >> 5;
  const int tile = blockIdx.x * 8 + wave;
  const int tm = tile >> 5;
  const int tn = tile & 31;
  const int row0 = tm * 16;
  const int col0 = tn * 16;

  const _Float16* abase = xh + (size_t)row0 * D;
  const _Float16* bbase = Wh + (size_t)col0 * D;

  v8f c = {};
  for (int k0 = 0; k0 < D; k0 += 32) {
    __builtin_prefetch(abase + k0 + 64, 0, 1);
    __builtin_prefetch(bbase + k0 + 64, 0, 1);
    v16h a = load_a_f16(abase + k0, D, lane);
    v16h b = load_b_f16(bbase + k0, D, lane);
    c = wmma_f16(a, b, c);
  }

  const int col = col0 + (lane & 15);
  const float bv = bias[col];
#pragma unroll
  for (int g = 0; g < 8; ++g) {
    const int row = row0 + g + ((lane >> 4) << 3);
    const float val = c[g] + bv;
    if (MODE == 0) {
      outh[(size_t)row * HD + col] = (_Float16)val;
    } else {
      const int bb = row >> 11;
      const int t  = row & (S - 1);
      outh[((size_t)bb * HD + col) * (size_t)S + t] = (_Float16)val;
    }
  }
}

// ---------------------------------------------------------------------------
// Squared row norms of q and k.
// ---------------------------------------------------------------------------
__global__ __launch_bounds__(256) void norms_kernel(
    const _Float16* __restrict__ qh, const _Float16* __restrict__ kh,
    float* __restrict__ qsq, float* __restrict__ ksq) {
  const int row = blockIdx.x;
  const _Float16* qp = qh + (size_t)row * HD;
  const _Float16* kp = kh + (size_t)row * HD;
  __shared__ float rq[256];
  __shared__ float rk[256];
  float aq = 0.f, ak = 0.f;
  for (int j = threadIdx.x; j < HD; j += 256) {
    float q = (float)qp[j]; aq += q * q;
    float k = (float)kp[j]; ak += k * k;
  }
  rq[threadIdx.x] = aq;
  rk[threadIdx.x] = ak;
  __syncthreads();
  for (int off = 128; off > 0; off >>= 1) {
    if (threadIdx.x < off) {
      rq[threadIdx.x] += rq[threadIdx.x + off];
      rk[threadIdx.x] += rk[threadIdx.x + off];
    }
    __syncthreads();
  }
  if (threadIdx.x == 0) {
    qsq[row] = rq[0];
    ksq[row] = rk[0];
  }
}

// ---------------------------------------------------------------------------
// Scores: workgroup computes a 128x128 block of attn per batch.
// TDM stages 128x32 q/k panels into double-buffered LDS; 8 waves each own a
// 32x64 sub-block (2x4 WMMA accumulators). Epilogue fuses
// exp(-max(sq,eps)/BW), stores to probs region, accumulates column sums.
// ---------------------------------------------------------------------------
__global__ __launch_bounds__(256) void scores_kernel(
    const _Float16* __restrict__ qh, const _Float16* __restrict__ kh,
    const float* __restrict__ qsq, const float* __restrict__ ksq,
    float* __restrict__ probsR, float* __restrict__ NC) {
  __shared__ _Float16 shA[2][128 * 32];
  __shared__ _Float16 shB[2][128 * 32];

  const int lane = threadIdx.x & 31;
  const int wave = threadIdx.x >> 5;
  const int blk  = blockIdx.x;          // B * 16 * 16 blocks
  const int b    = blk >> 8;
  const int bi   = (blk >> 4) & 15;
  const int bj   = blk & 15;
  const int row0 = bi * 128;            // within batch
  const int col0 = bj * 128;

  const int rg = wave & 3;              // 32-row group
  const int cg = wave >> 2;             // 64-col group

  const u64 qbase = (u64)(uintptr_t)(qh + (size_t)(b * S + row0) * HD);
  const u64 kbase = (u64)(uintptr_t)(kh + (size_t)(b * S + col0) * HD);
  const u32 ldsA0 = (u32)(uintptr_t)(&shA[0][0]);
  const u32 ldsA1 = (u32)(uintptr_t)(&shA[1][0]);
  const u32 ldsB0 = (u32)(uintptr_t)(&shB[0][0]);
  const u32 ldsB1 = (u32)(uintptr_t)(&shB[1][0]);

  constexpr int NITER = HD / 32;        // 16 k-steps

  if (wave == 0) {                      // prologue: fill buffer 0
    tdm_load_2d_f16(ldsA0, qbase, HD, 128, 32, 128, HD);
    tdm_load_2d_f16(ldsB0, kbase, HD, 128, 32, 128, HD);
  }

  v8f c[2][4] = {};
  for (int i = 0; i < NITER; ++i) {
    const int cur = i & 1;
    if (wave == 0) {
      if (i + 1 < NITER) {
        const u64 koff = (u64)(i + 1) * 32 * 2;   // bytes along K
        tdm_load_2d_f16(cur ? ldsA0 : ldsA1, qbase + koff, HD, 128, 32, 128, HD);
        tdm_load_2d_f16(cur ? ldsB0 : ldsB1, kbase + koff, HD, 128, 32, 128, HD);
        __builtin_amdgcn_s_wait_tensorcnt(2);     // current buffer complete
      } else {
        __builtin_amdgcn_s_wait_tensorcnt(0);
      }
    }
    __syncthreads();                              // publish buffer `cur`

    const _Float16* pA = &shA[cur][0];
    const _Float16* pB = &shB[cur][0];
    v16h a[2], bm[4];
#pragma unroll
    for (int mi = 0; mi < 2; ++mi)
      a[mi] = load_a_f16(pA + (rg * 32 + mi * 16) * 32, 32, lane);
#pragma unroll
    for (int ni = 0; ni < 4; ++ni)
      bm[ni] = load_b_f16(pB + (cg * 64 + ni * 16) * 32, 32, lane);
#pragma unroll
    for (int mi = 0; mi < 2; ++mi)
#pragma unroll
      for (int ni = 0; ni < 4; ++ni)
        c[mi][ni] = wmma_f16(a[mi], bm[ni], c[mi][ni]);

    __syncthreads();                              // readers done before overwrite
  }

  // Epilogue: sq -> attn -> store + column sums.
#pragma unroll
  for (int ni = 0; ni < 4; ++ni) {
    const int t  = col0 + cg * 64 + ni * 16 + (lane & 15);
    const float kk = ksq[b * S + t];
    float cs = 0.f;
#pragma unroll
    for (int mi = 0; mi < 2; ++mi) {
#pragma unroll
      for (int g = 0; g < 8; ++g) {
        const int s = row0 + rg * 32 + mi * 16 + g + ((lane >> 4) << 3);
        const float sq = qsq[b * S + s] + kk - 2.0f * c[mi][ni][g];
        const float at = __expf(-fmaxf(sq, EPS) * INV_BW);
        probsR[(size_t)(b * S + s) * S + t] = at;
        cs += at;
      }
    }
    atomicAdd(&NC[b * S + t], cs);
  }
}

// ---------------------------------------------------------------------------
// Per-row normalize. N_R^{-a} cancels in the p=1 normalization:
// probs(s,t) = attn(s,t)*NC(t)^{-1/2} / sum_t attn(s,t)*NC(t)^{-1/2}.
// ---------------------------------------------------------------------------
__global__ __launch_bounds__(256) void normalize_kernel(
    float* __restrict__ probsR, const float* __restrict__ NC) {
  const int row = blockIdx.x;
  const int b   = row >> 11;
  float* pr = probsR + (size_t)row * S;
  const float* nc = NC + (size_t)b * S;
  __shared__ float red[256];
  float part = 0.f;
  for (int t = threadIdx.x; t < S; t += 256) {
    float w = pr[t] * rsqrtf(nc[t]);
    pr[t] = w;
    part += w;
  }
  red[threadIdx.x] = part;
  __syncthreads();
  for (int off = 128; off > 0; off >>= 1) {
    if (threadIdx.x < off) red[threadIdx.x] += red[threadIdx.x + off];
    __syncthreads();
  }
  const float inv = 1.0f / fmaxf(red[0], EPS);
  for (int t = threadIdx.x; t < S; t += 256) pr[t] *= inv;
}

// ---------------------------------------------------------------------------
// out = probs @ v (per batch). Register-blocked: each wave computes 32x64
// (2x4 WMMA accumulators); probs converted f32->f16 on load.
// ---------------------------------------------------------------------------
__global__ __launch_bounds__(256) void out_kernel(
    const float* __restrict__ probsR, const _Float16* __restrict__ vT,
    float* __restrict__ outp) {
  const int lane = threadIdx.x & 31;
  const int wave = threadIdx.x >> 5;
  const int job  = blockIdx.x * 8 + wave;       // (B*S/32)*(HD/64) = 2048
  const int tm = job >> 3;                      // 0..255 (32-row strips)
  const int tn = job & 7;                       // 0..7   (64-col strips)
  const int b  = tm >> 6;
  const int s0 = (tm & 63) * 32;
  const int col0 = tn * 64;

  const float*    abase = probsR + (size_t)(b * S + s0) * S;
  const _Float16* bbase = vT + ((size_t)b * HD + col0) * (size_t)S;

  v8f c[2][4] = {};
  for (int k0 = 0; k0 < S; k0 += 32) {
    __builtin_prefetch(abase + k0 + 64, 0, 1);
    __builtin_prefetch(bbase + k0 + 64, 0, 1);
    v16h a[2], bm[4];
#pragma unroll
    for (int mi = 0; mi < 2; ++mi)
      a[mi] = load_a_f32(abase + (size_t)(mi * 16) * S + k0, S, lane);
#pragma unroll
    for (int ni = 0; ni < 4; ++ni)
      bm[ni] = load_b_f16(bbase + (size_t)(ni * 16) * S + k0, S, lane);
#pragma unroll
    for (int mi = 0; mi < 2; ++mi)
#pragma unroll
      for (int ni = 0; ni < 4; ++ni)
        c[mi][ni] = wmma_f16(a[mi], bm[ni], c[mi][ni]);
  }

#pragma unroll
  for (int mi = 0; mi < 2; ++mi) {
#pragma unroll
    for (int g = 0; g < 8; ++g) {
      const int s = s0 + mi * 16 + g + ((lane >> 4) << 3);
#pragma unroll
      for (int ni = 0; ni < 4; ++ni) {
        outp[(size_t)(b * S + s) * HD + col0 + ni * 16 + (lane & 15)] =
            c[mi][ni][g];
      }
    }
  }
}

// ---------------------------------------------------------------------------
extern "C" void kernel_launch(void* const* d_in, const int* in_sizes, int n_in,
                              void* d_out, int out_size, void* d_ws, size_t ws_size,
                              hipStream_t stream) {
  const float* x  = (const float*)d_in[0];
  const float* Wq = (const float*)d_in[1];
  const float* bq = (const float*)d_in[2];
  const float* Wk = (const float*)d_in[3];
  const float* bk = (const float*)d_in[4];
  const float* Wv = (const float*)d_in[5];
  const float* bv = (const float*)d_in[6];

  float* outp   = (float*)d_out;                       // (B,1,S,HD)
  float* probsR = outp + (size_t)B * S * HD;           // (B,1,S,S)

  char* ws = (char*)d_ws;
  size_t off = 0;
  auto carve = [&](size_t bytes) -> void* {
    void* p = ws + off;
    off += (bytes + 255) & ~(size_t)255;
    return p;
  };
  _Float16* xh  = (_Float16*)carve((size_t)B * S * D * 2);
  _Float16* Wqh = (_Float16*)carve((size_t)HD * D * 2);
  _Float16* Wkh = (_Float16*)carve((size_t)HD * D * 2);
  _Float16* Wvh = (_Float16*)carve((size_t)HD * D * 2);
  _Float16* qh  = (_Float16*)carve((size_t)B * S * HD * 2);
  _Float16* kh  = (_Float16*)carve((size_t)B * S * HD * 2);
  _Float16* vT  = (_Float16*)carve((size_t)B * HD * S * 2);
  float*    qsq = (float*)carve((size_t)B * S * 4);
  float*    ksq = (float*)carve((size_t)B * S * 4);
  float*    NC  = (float*)carve((size_t)B * S * 4);
  (void)ws_size; (void)in_sizes; (void)n_in; (void)out_size;

  // 1) f32 -> f16 conversions + zero column sums
  {
    int n = B * S * D;
    cvt_f32_f16_kernel<<<(n + 255) / 256, 256, 0, stream>>>(x, xh, n);
    int nw = HD * D;
    cvt_f32_f16_kernel<<<(nw + 255) / 256, 256, 0, stream>>>(Wq, Wqh, nw);
    cvt_f32_f16_kernel<<<(nw + 255) / 256, 256, 0, stream>>>(Wk, Wkh, nw);
    cvt_f32_f16_kernel<<<(nw + 255) / 256, 256, 0, stream>>>(Wv, Wvh, nw);
    int nc = B * S;
    zero_f32_kernel<<<(nc + 255) / 256, 256, 0, stream>>>(NC, nc);
  }

  // 2) projections (WMMA): q, k row-major; v transposed per batch
  {
    const int tiles = (B * S / 16) * (HD / 16);          // 16384
    proj_kernel<0><<<tiles / 8, 256, 0, stream>>>(xh, Wqh, bq, qh);
    proj_kernel<0><<<tiles / 8, 256, 0, stream>>>(xh, Wkh, bk, kh);
    proj_kernel<1><<<tiles / 8, 256, 0, stream>>>(xh, Wvh, bv, vT);
  }

  // 3) row norms
  norms_kernel<<<B * S, 256, 0, stream>>>(qh, kh, qsq, ksq);

  // 4) scores + column sums (TDM-staged WMMA, 128x128 blocks)
  scores_kernel<<<B * 16 * 16, 256, 0, stream>>>(qh, kh, qsq, ksq, probsR, NC);

  // 5) bistochastic scaling + p=1 row-normalize (in place in d_out)
  normalize_kernel<<<B * S, 256, 0, stream>>>(probsR, NC);

  // 6) out = probs @ v (WMMA, register-blocked)
  out_kernel<<<(B * S / 32) * (HD / 64) / 8, 256, 0, stream>>>(probsR, vT, outp);
}
